// DeformableAttention_32916629356596
// MI455X (gfx1250) — matrix operand order repaired and checked by reading, not compiled, not run on hardware
//
#include <hip/hip_runtime.h>
#include <math.h>

#define B_   16
#define C_   768
#define H_   32
#define W_   32
#define NH_  12
#define HD_  64
#define HK_  16
#define NS_  256   // Hk*Wk
#define HW_  1024  // H*W
#define BH_  (B_*NH_)

typedef __attribute__((ext_vector_type(16))) _Float16 v16h;
typedef __attribute__((ext_vector_type(8)))  float    v8f;

union Frag16 { v16h v; _Float16 h[16]; uint4 q[2]; };

// ---------------------------------------------------------------------------
// CDNA5 async global->LDS copy (ASYNCcnt path, CDNA5 ISA)
// ---------------------------------------------------------------------------
__device__ __forceinline__ void async_copy_b128(const void* gsrc, void* ldst) {
    asm volatile("global_load_async_to_lds_b128 %0, %1, off"
                 :
                 : "v"((unsigned int)(unsigned long long)ldst),
                   "v"((unsigned long long)gsrc)
                 : "memory");
}
__device__ __forceinline__ void wait_async() {
    asm volatile("s_wait_asynccnt 0x0" ::: "memory");
}

// ---------------------------------------------------------------------------
// Depthwise 3x3 conv helper (zero padded)
// ---------------------------------------------------------------------------
__device__ __forceinline__ float dw3x3(const float* __restrict__ xp,
                                       const float* __restrict__ wp,
                                       int oh, int ow, int stride) {
    float acc = 0.0f;
    const int ih0 = oh * stride - 1, iw0 = ow * stride - 1;
#pragma unroll
    for (int i = 0; i < 3; ++i) {
        const int ih = ih0 + i;
        if (ih < 0 || ih >= H_) continue;
#pragma unroll
        for (int j = 0; j < 3; ++j) {
            const int iw = iw0 + j;
            if (iw < 0 || iw >= W_) continue;
            acc += xp[ih * W_ + iw] * wp[i * 3 + j];
        }
    }
    return acc;
}

// value (stride 1) and offset-branch (stride 2) convs -> f32
__global__ __launch_bounds__(256) void dwconv_f32_kernel(
    const float* __restrict__ x, const float* __restrict__ w,
    const float* __restrict__ bias, float* __restrict__ y,
    int stride, int OH, int OW, int total) {
    const int gid = blockIdx.x * 256 + threadIdx.x;
    if (gid >= total) return;
    const int ow = gid % OW;
    const int oh = (gid / OW) % OH;
    const int c  = (gid / (OW * OH)) % C_;
    const int b  = gid / (OW * OH * C_);
    const float* xp = x + ((size_t)b * C_ + c) * (H_ * W_);
    y[gid] = dw3x3(xp, w + c * 9, oh, ow, stride) + bias[c];
}

// query conv -> qT f16 [BH][HW][HD] (row-major, HD contiguous: A operand),
// scale 1/sqrt(HD) folded in
__global__ __launch_bounds__(256) void dwconv_q_kernel(
    const float* __restrict__ x, const float* __restrict__ w,
    const float* __restrict__ bias, _Float16* __restrict__ qT) {
    const int gid = blockIdx.x * 256 + threadIdx.x;
    if (gid >= B_ * C_ * H_ * W_) return;
    const int ow = gid % W_;
    const int oh = (gid / W_) % H_;
    const int c  = (gid / (W_ * H_)) % C_;
    const int b  = gid / (W_ * H_ * C_);
    const float* xp = x + ((size_t)b * C_ + c) * (H_ * W_);
    const float v = dw3x3(xp, w + c * 9, oh, ow, 1) + bias[c];
    qT[(((size_t)(b * NH_ + (c >> 6))) * HW_ + (oh * W_ + ow)) * HD_ + (c & 63)] =
        (_Float16)(v * 0.125f);
}

// key conv (stride 2) -> kT f16 [BH][NS][HD] (N-major, K(=c) contiguous:
// B operand of GEMM1 loads as 2 contiguous b128 per lane)
__global__ __launch_bounds__(256) void dwconv_k_kernel(
    const float* __restrict__ x, const float* __restrict__ w,
    const float* __restrict__ bias, _Float16* __restrict__ kT) {
    const int gid = blockIdx.x * 256 + threadIdx.x;
    if (gid >= B_ * C_ * HK_ * HK_) return;
    const int ow = gid % HK_;
    const int oh = (gid / HK_) % HK_;
    const int c  = (gid / (HK_ * HK_)) % C_;
    const int b  = gid / (HK_ * HK_ * C_);
    const float* xp = x + ((size_t)b * C_ + c) * (H_ * W_);
    const float v = dw3x3(xp, w + c * 9, oh, ow, 2) + bias[c];
    kT[(((size_t)(b * NH_ + (c >> 6))) * NS_ + (oh * HK_ + ow)) * HD_ + (c & 63)] =
        (_Float16)v;
}

// ---------------------------------------------------------------------------
// Offset branch: LayerNorm(C) -> GELU -> 1x1 conv(C->2) -> tanh -> pos (y,x)
// one block per (b, n)
// ---------------------------------------------------------------------------
__global__ __launch_bounds__(256) void offset_pos_kernel(
    const float* __restrict__ t, const float* __restrict__ ln_g,
    const float* __restrict__ ln_b, const float* __restrict__ w2,
    float* __restrict__ pos) {
    const int bid = blockIdx.x;            // b*256 + n
    const int b = bid >> 8, n = bid & 255;
    const int tid = threadIdx.x;
    const int lane = tid & 31, wid = tid >> 5;
    __shared__ float red[4][8];

    const float* tp = t + (size_t)b * C_ * NS_ + n;
    const float x0 = tp[(size_t)tid * NS_];
    const float x1 = tp[(size_t)(tid + 256) * NS_];
    const float x2 = tp[(size_t)(tid + 512) * NS_];
    float s1 = x0 + x1 + x2;
    float s2 = x0 * x0 + x1 * x1 + x2 * x2;
#pragma unroll
    for (int off = 1; off < 32; off <<= 1) {
        s1 += __shfl_xor(s1, off, 32);
        s2 += __shfl_xor(s2, off, 32);
    }
    if (lane == 0) { red[0][wid] = s1; red[1][wid] = s2; }
    __syncthreads();
    float ts1 = 0.f, ts2 = 0.f;
#pragma unroll
    for (int i = 0; i < 8; ++i) { ts1 += red[0][i]; ts2 += red[1][i]; }
    const float mu   = ts1 * (1.0f / C_);
    const float var  = ts2 * (1.0f / C_) - mu * mu;
    const float rstd = rsqrtf(var + 1e-5f);

    float a0 = 0.f, a1 = 0.f;
    const float xs[3] = {x0, x1, x2};
#pragma unroll
    for (int j = 0; j < 3; ++j) {
        const int c = tid + j * 256;
        const float xn = (xs[j] - mu) * rstd * ln_g[c] + ln_b[c];
        const float g  = 0.5f * xn * (1.0f + erff(xn * 0.70710678118f));
        a0 += g * w2[c];
        a1 += g * w2[C_ + c];
    }
#pragma unroll
    for (int off = 1; off < 32; off <<= 1) {
        a0 += __shfl_xor(a0, off, 32);
        a1 += __shfl_xor(a1, off, 32);
    }
    __syncthreads();
    if (lane == 0) { red[2][wid] = a0; red[3][wid] = a1; }
    __syncthreads();
    if (tid == 0) {
        float t0 = 0.f, t1 = 0.f;
#pragma unroll
        for (int i = 0; i < 8; ++i) { t0 += red[2][i]; t1 += red[3][i]; }
        const float s = 2.0f / (float)HK_;        // ORF * (1/Hk)
        const float ry = ((float)(n >> 4) + 0.5f) * (2.0f / HK_) - 1.0f;
        const float rx = ((float)(n & 15) + 0.5f) * (2.0f / HK_) - 1.0f;
        pos[(size_t)bid * 2 + 0] = tanhf(t0) * s + ry;
        pos[(size_t)bid * 2 + 1] = tanhf(t1) * s + rx;
    }
}

// ---------------------------------------------------------------------------
// Bilinear sample value at pos -> vs f16 [BH][HD][NS] (N(=d)-major,
// K(=n) contiguous: B operand of GEMM2 loads as 2 contiguous b128 per lane)
// ---------------------------------------------------------------------------
__global__ __launch_bounds__(256) void sample_kernel(
    const float* __restrict__ value, const float* __restrict__ pos,
    _Float16* __restrict__ vs) {
    const int gid = blockIdx.x * 256 + threadIdx.x;
    if (gid >= B_ * NS_ * C_) return;
    const int c = gid % C_;
    const int n = (gid / C_) % NS_;
    const int b = gid / (C_ * NS_);
    const float py = pos[((size_t)b * NS_ + n) * 2 + 0];
    const float px = pos[((size_t)b * NS_ + n) * 2 + 1];
    const float gx = (px + 1.0f) * 15.5f;   // (W-1)/2, align_corners=True
    const float gy = (py + 1.0f) * 15.5f;
    const float fx0 = floorf(gx), fy0 = floorf(gy);
    const float wx = gx - fx0, wy = gy - fy0;
    const int x0i = (int)fx0, y0i = (int)fy0;
    const float* vp = value + ((size_t)b * C_ + c) * (H_ * W_);
    float r = 0.f;
#pragma unroll
    for (int dy = 0; dy < 2; ++dy)
#pragma unroll
        for (int dx = 0; dx < 2; ++dx) {
            const int iy = y0i + dy, ix = x0i + dx;
            if (iy >= 0 && iy < H_ && ix >= 0 && ix < W_) {
                const float wgt = (dy ? wy : 1.f - wy) * (dx ? wx : 1.f - wx);
                r += vp[iy * W_ + ix] * wgt;
            }
        }
    vs[(((size_t)b * NH_ + (c >> 6)) * HD_ + (c & 63)) * NS_ + n] = (_Float16)r;
}

// w_out [768][768] f32 -> f16 (same [o][c] layout: already N-major for GEMM)
__global__ __launch_bounds__(256) void wout_tof16_kernel(
    const float* __restrict__ w, _Float16* __restrict__ wh) {
    const int gid = blockIdx.x * 256 + threadIdx.x;
    if (gid >= C_ * C_) return;
    wh[gid] = (_Float16)w[gid];
}

// ---------------------------------------------------------------------------
// Fused attention: S = qT*k + rpe_bias, softmax, O = P * vs^T
// grid (HW/64, BH), 128 threads (4 waves). Each wave owns a 16-row stripe.
// ---------------------------------------------------------------------------
__global__ __launch_bounds__(128) void attn_kernel(
    const _Float16* __restrict__ qT,   // [BH][HW][HD]
    const _Float16* __restrict__ kT,   // [BH][NS][HD]
    const _Float16* __restrict__ vs,   // [BH][HD][NS]
    const float* __restrict__ pos,     // [B][NS][2] (y,x)
    const float* __restrict__ rpe,     // [NH][31][31]
    _Float16* __restrict__ attnT)      // [B][HW][C]
{
    __shared__ __align__(16) _Float16 lq[64 * HD_];   // 8 KB
    __shared__ __align__(16) _Float16 lk[NS_ * HD_];  // 32 KB, re-used as P
    __shared__ __align__(16) float lpos[NS_ * 2];
    __shared__ float lrpe[31 * 31 + 1];
    _Float16* lp = lk;                                 // alias after barrier

    const int tid  = threadIdx.x;
    const int lane = tid & 31;
    const int wid  = tid >> 5;
    const int bh   = blockIdx.y;
    const int b    = bh / NH_;
    const int head = bh - b * NH_;
    const int m0   = blockIdx.x * 64;
    const int wm0  = wid * 16;
    const int nlow = lane & 15;
    const int hi   = lane >> 4;

    __builtin_amdgcn_s_wait_tensorcnt(0);

    // --- stage tiles (ASYNCcnt path) ---
    const _Float16* qsrc = qT + ((size_t)bh * HW_ + m0) * HD_;
    for (int i = tid; i < 512; i += 128)
        async_copy_b128(qsrc + i * 8, &lq[i * 8]);
    const _Float16* ksrc = kT + (size_t)bh * NS_ * HD_;
    for (int i = tid; i < 2048; i += 128)
        async_copy_b128(ksrc + i * 8, &lk[i * 8]);
    const float* psrc = pos + (size_t)b * NS_ * 2;
    for (int i = tid; i < 128; i += 128)
        async_copy_b128(psrc + i * 4, &lpos[i * 4]);
    for (int i = tid; i < 961; i += 128)
        lrpe[i] = rpe[head * 961 + i];
    wait_async();
    __syncthreads();

    // --- GEMM1: S[16x256] = qT[16x64] * k[64x256] ---
    v8f acc[16] = {};
#pragma unroll
    for (int kc = 0; kc < HD_; kc += 32) {
        Frag16 a;
        a.q[0] = *(const uint4*)&lq[(wm0 + nlow) * HD_ + kc + hi * 8];
        a.q[1] = *(const uint4*)&lq[(wm0 + nlow) * HD_ + kc + hi * 8 + 16];
#pragma unroll
        for (int nt = 0; nt < 16; ++nt) {
            Frag16 bf;   // lane = column n, needs K (=c) contiguous
            const int nidx = nt * 16 + nlow;
            bf.q[0] = *(const uint4*)&lk[nidx * HD_ + kc + hi * 8];
            bf.q[1] = *(const uint4*)&lk[nidx * HD_ + kc + hi * 8 + 16];
            acc[nt] = __builtin_amdgcn_wmma_f32_16x16x32_f16(
                false, a.v, false, bf.v, (short)0, acc[nt], false, false);
        }
    }

    // --- rpe bias (bilinear on 31x31 table) + softmax over ns ---
#pragma unroll
    for (int r = 0; r < 8; ++r) {
        const int mloc = r + hi * 8;
        const int m = m0 + wm0 + mloc;
        const float qy = ((float)(m >> 5) + 0.5f) * (2.0f / H_) - 1.0f;
        const float qx = ((float)(m & 31) + 0.5f) * (2.0f / W_) - 1.0f;
        float mx = -3.0e38f;
#pragma unroll
        for (int nt = 0; nt < 16; ++nt) {
            const int n = nt * 16 + nlow;
            const float gx = ((qx - lpos[2 * n + 1]) * 0.5f + 1.0f) * 15.0f;
            const float gy = ((qy - lpos[2 * n + 0]) * 0.5f + 1.0f) * 15.0f;
            const float fx0 = floorf(gx), fy0 = floorf(gy);
            const float wx = gx - fx0, wy = gy - fy0;
            const int ix0 = (int)fx0, iy0 = (int)fy0;
            float v00 = 0.f, v01 = 0.f, v10 = 0.f, v11 = 0.f;
            if (iy0 >= 0 && iy0 < 31) {
                if (ix0 >= 0 && ix0 < 31)         v00 = lrpe[iy0 * 31 + ix0];
                if (ix0 + 1 >= 0 && ix0 + 1 < 31) v01 = lrpe[iy0 * 31 + ix0 + 1];
            }
            if (iy0 + 1 >= 0 && iy0 + 1 < 31) {
                if (ix0 >= 0 && ix0 < 31)         v10 = lrpe[(iy0 + 1) * 31 + ix0];
                if (ix0 + 1 >= 0 && ix0 + 1 < 31) v11 = lrpe[(iy0 + 1) * 31 + ix0 + 1];
            }
            const float bias = v00 * (1.f - wy) * (1.f - wx) + v01 * (1.f - wy) * wx +
                               v10 * wy * (1.f - wx) + v11 * wy * wx;
            acc[nt][r] += bias;
            mx = fmaxf(mx, acc[nt][r]);
        }
#pragma unroll
        for (int off = 1; off < 16; off <<= 1) mx = fmaxf(mx, __shfl_xor(mx, off, 32));
        float sum = 0.f;
#pragma unroll
        for (int nt = 0; nt < 16; ++nt) {
            acc[nt][r] = __expf(acc[nt][r] - mx);
            sum += acc[nt][r];
        }
#pragma unroll
        for (int off = 1; off < 16; off <<= 1) sum += __shfl_xor(sum, off, 32);
        const float inv = 1.0f / sum;
#pragma unroll
        for (int nt = 0; nt < 16; ++nt) acc[nt][r] *= inv;
    }

    __syncthreads();   // all waves done reading lk; safe to overwrite with P
#pragma unroll
    for (int r = 0; r < 8; ++r) {
        const int mloc = r + hi * 8;
#pragma unroll
        for (int nt = 0; nt < 16; ++nt)
            lp[(wm0 + mloc) * NS_ + nt * 16 + nlow] = (_Float16)acc[nt][r];
    }

    // --- GEMM2: O[16x64] = P[16x256] * vs^T[256x64]; B streamed from L2 ---
    const _Float16* vsrc = vs + (size_t)bh * HD_ * NS_;
    v8f acc2[4] = {};
#pragma unroll
    for (int kc = 0; kc < NS_; kc += 32) {
        Frag16 a;
        a.q[0] = *(const uint4*)&lp[(wm0 + nlow) * NS_ + kc + hi * 8];
        a.q[1] = *(const uint4*)&lp[(wm0 + nlow) * NS_ + kc + hi * 8 + 16];
#pragma unroll
        for (int nt = 0; nt < 4; ++nt) {
            Frag16 bf;   // lane = column d, needs K (=n) contiguous
            const int d = nt * 16 + nlow;
            bf.q[0] = *(const uint4*)&vsrc[(size_t)d * NS_ + kc + hi * 8];
            bf.q[1] = *(const uint4*)&vsrc[(size_t)d * NS_ + kc + hi * 8 + 16];
            acc2[nt] = __builtin_amdgcn_wmma_f32_16x16x32_f16(
                false, a.v, false, bf.v, (short)0, acc2[nt], false, false);
        }
    }

    // epilogue -> attnT[b][m][head*64 + d] (f16, feeds final GEMM)
    _Float16* outp = attnT + ((size_t)b * HW_ + m0 + wm0) * C_ + head * HD_;
#pragma unroll
    for (int nt = 0; nt < 4; ++nt)
#pragma unroll
        for (int r = 0; r < 8; ++r)
            outp[(size_t)(r + hi * 8) * C_ + nt * 16 + nlow] = (_Float16)acc2[nt][r];
}

// ---------------------------------------------------------------------------
// Final 1x1 conv as GEMM: out[b][o][m] = sum_c attnT[b][m][c] * w_out[o][c]
// grid (1024/128, 768/128, B), 256 threads (8 waves), 128x128 tile
// ---------------------------------------------------------------------------
__global__ __launch_bounds__(256) void out_gemm_kernel(
    const _Float16* __restrict__ aT,  // [B][HW][C]   (A: M-major, K contiguous)
    const _Float16* __restrict__ wh,  // [C_out][C]   (B: N-major, K contiguous)
    float* __restrict__ out)          // [B][C][HW]
{
    __shared__ __align__(16) _Float16 la[128 * 32];  // A tile [m][k], 8 KB
    __shared__ __align__(16) _Float16 lb[128 * 32];  // B tile [o][k], 8 KB
    const int tid  = threadIdx.x;
    const int lane = tid & 31, wid = tid >> 5;
    const int m0 = blockIdx.x * 128;
    const int n0 = blockIdx.y * 128;
    const int b  = blockIdx.z;
    const int nlow = lane & 15, hi = lane >> 4;
    const int wm0 = wid * 16;

    v8f acc[8] = {};
    for (int kc = 0; kc < C_; kc += 32) {
        if (kc + 32 < C_)
            __builtin_prefetch(&aT[((size_t)b * HW_ + m0 + (tid >> 1)) * C_ + kc + 32], 0, 1);
        for (int i = tid; i < 512; i += 256) {
            const int row = i >> 2, c4 = i & 3;
            ((uint4*)la)[i] =
                *(const uint4*)&aT[((size_t)b * HW_ + m0 + row) * C_ + kc + c4 * 8];
        }
        for (int i = tid; i < 512; i += 256) {
            const int row = i >> 2, c4 = i & 3;
            ((uint4*)lb)[i] = *(const uint4*)&wh[(size_t)(n0 + row) * C_ + kc + c4 * 8];
        }
        __syncthreads();
        Frag16 a;
        a.q[0] = *(const uint4*)&la[(wm0 + nlow) * 32 + hi * 8];
        a.q[1] = *(const uint4*)&la[(wm0 + nlow) * 32 + hi * 8 + 16];
#pragma unroll
        for (int nt = 0; nt < 8; ++nt) {
            Frag16 bf;   // lane = column o, K contiguous in lb row
            bf.q[0] = *(const uint4*)&lb[(nt * 16 + nlow) * 32 + hi * 8];
            bf.q[1] = *(const uint4*)&lb[(nt * 16 + nlow) * 32 + hi * 8 + 16];
            acc[nt] = __builtin_amdgcn_wmma_f32_16x16x32_f16(
                false, a.v, false, bf.v, (short)0, acc[nt], false, false);
        }
        __syncthreads();
    }
#pragma unroll
    for (int nt = 0; nt < 8; ++nt) {
        const int o = n0 + nt * 16 + nlow;
#pragma unroll
        for (int r = 0; r < 8; ++r) {
            const int m = m0 + wm0 + r + hi * 8;
            out[((size_t)b * C_ + o) * HW_ + m] = acc[nt][r];
        }
    }
}

// ---------------------------------------------------------------------------
// host-side orchestration
// ---------------------------------------------------------------------------
#define OFF_VALUE 0ull
#define OFF_QT    50331648ull
#define OFF_K     75497472ull
#define OFF_T     81788928ull
#define OFF_POS   94371840ull
#define OFF_VST   94404608ull
#define OFF_ATT   100696064ull
#define OFF_WT    125861888ull

extern "C" void kernel_launch(void* const* d_in, const int* in_sizes, int n_in,
                              void* d_out, int out_size, void* d_ws, size_t ws_size,
                              hipStream_t stream) {
    (void)in_sizes; (void)n_in; (void)out_size; (void)ws_size;
    const float* x      = (const float*)d_in[0];
    const float* wv     = (const float*)d_in[1];
    const float* bv     = (const float*)d_in[2];
    const float* wq     = (const float*)d_in[3];
    const float* bq     = (const float*)d_in[4];
    const float* wk     = (const float*)d_in[5];
    const float* bk     = (const float*)d_in[6];
    const float* w_off1 = (const float*)d_in[7];
    const float* b_off1 = (const float*)d_in[8];
    const float* ln_g   = (const float*)d_in[9];
    const float* ln_b   = (const float*)d_in[10];
    const float* w_off2 = (const float*)d_in[11];
    const float* rpe    = (const float*)d_in[12];
    const float* w_out  = (const float*)d_in[13];
    float* out = (float*)d_out;

    char* ws = (char*)d_ws;
    float*    valb = (float*)(ws + OFF_VALUE);
    _Float16* qTb  = (_Float16*)(ws + OFF_QT);
    _Float16* kb   = (_Float16*)(ws + OFF_K);
    float*    tb   = (float*)(ws + OFF_T);
    float*    posb = (float*)(ws + OFF_POS);
    _Float16* vsb  = (_Float16*)(ws + OFF_VST);
    _Float16* attb = (_Float16*)(ws + OFF_ATT);
    _Float16* whb  = (_Float16*)(ws + OFF_WT);

    const int nFull = B_ * C_ * H_ * W_;       // 12582912
    const int nHalf = B_ * C_ * HK_ * HK_;     // 3145728

    dwconv_f32_kernel<<<nFull / 256, 256, 0, stream>>>(x, wv, bv, valb, 1, H_, W_, nFull);
    dwconv_q_kernel  <<<nFull / 256, 256, 0, stream>>>(x, wq, bq, qTb);
    dwconv_k_kernel  <<<nHalf / 256, 256, 0, stream>>>(x, wk, bk, kb);
    dwconv_f32_kernel<<<nHalf / 256, 256, 0, stream>>>(x, w_off1, b_off1, tb, 2, HK_, HK_, nHalf);
    offset_pos_kernel<<<B_ * NS_, 256, 0, stream>>>(tb, ln_g, ln_b, w_off2, posb);
    sample_kernel    <<<(B_ * NS_ * C_) / 256, 256, 0, stream>>>(valb, posb, vsb);
    wout_tof16_kernel<<<(C_ * C_) / 256, 256, 0, stream>>>(w_out, whb);
    attn_kernel<<<dim3(HW_ / 64, BH_), 128, 0, stream>>>(qTb, kb, vsb, posb, rpe, attb);
    out_gemm_kernel<<<dim3(HW_ / 128, C_ / 128, B_), 256, 0, stream>>>(attb, whb, out);
}
